// LGGPT_14929306321003
// MI455X (gfx1250) — compile-verified, matrix operand order checked by reading, not run on hardware
//
#include <hip/hip_runtime.h>
#include <hip/hip_bf16.h>

// ---------------------------------------------------------------------------
// GPT forward (B=2,T=2048,E=1024,H=16,HD=64,V=32000,L=3) for gfx1250 (MI455X)
// f16 WMMA (v_wmma_f32_16x16x32_f16), fp32 accumulation/state, TDM staging.
// ---------------------------------------------------------------------------

typedef __attribute__((ext_vector_type(16))) _Float16 v16h;
typedef __attribute__((ext_vector_type(8)))  float    v8f;
typedef __attribute__((ext_vector_type(4)))  unsigned int v4u;
typedef __attribute__((ext_vector_type(8)))  int      v8i;
typedef __attribute__((ext_vector_type(4)))  int      v4i;

#define B_   2
#define T_   2048
#define E_   1024
#define H_   16
#define HD_  64
#define V_   32000
#define L_   3
#define FF_  4096
#define NTOK (B_*T_)

#if defined(__gfx1250__) && __has_builtin(__builtin_amdgcn_tensor_load_to_lds)
#define USE_TDM 1
#else
#define USE_TDM 0
#endif

union FragU { v16h v; unsigned int u[8]; };

// A-style fragment load (also used for B staged as [N][K] in LDS).
// Lane L holds row (L&15); half j-pair covers K = (j<4?0:16) + 8*(L>>4) + 2*(j&3)+p.
__device__ __forceinline__ v16h load_frag(const _Float16* rp, int lane) {
  const int g8 = (lane & 16) >> 1;           // 0 or 8
  FragU f;
#pragma unroll
  for (int j = 0; j < 8; ++j) {
    const int kk = ((j & 4) << 2) + g8 + ((j & 3) << 1);
    f.u[j] = *(const unsigned int*)(rp + kk);
  }
  return f.v;
}

__device__ __forceinline__ float hmax16(float x) {
#pragma unroll
  for (int m = 1; m < 16; m <<= 1) x = fmaxf(x, __shfl_xor(x, m, 32));
  return x;
}
__device__ __forceinline__ float hsum16(float x) {
#pragma unroll
  for (int m = 1; m < 16; m <<= 1) x += __shfl_xor(x, m, 32);
  return x;
}

#if USE_TDM
// Issue a TDM 2D tile load: tile 32(K) x 128(M) f16 rows from src (row pitch
// rowStrideElems) into LDS at lds_off with pitch-40-halves padding
// (pad after every 16 DWORDs (code 3) by 4 DWORDs (code 3)).
__device__ __forceinline__ void tdm_load_tile_a(const _Float16* src,
                                                unsigned lds_off,
                                                int rowStrideElems,
                                                int tensorW, int tensorH) {
  const unsigned long long ga = (unsigned long long)(const void*)src;
  v4u g0;
  g0.x = 1u;                                        // count=1, user descriptor
  g0.y = lds_off;                                   // LDS byte address
  g0.z = (unsigned)(ga & 0xFFFFFFFFull);            // global_addr[31:0]
  g0.w = (unsigned)((ga >> 32) & 0x1FFFFFFull)      // global_addr[56:32]
         | (2u << 30);                              // type=2 ("image")
  v8i g1;
  g1[0] = (int)((1u << 16)                          // data_size=1 -> 2 bytes
              | (1u << 20)                          // pad_enable
              | (3u << 22)                          // pad_interval: 16 DWORDs
              | (3u << 25));                        // pad_amount: 4 DWORDs
  g1[1] = (int)(((unsigned)tensorW & 0xFFFFu) << 16);           // tensor_dim0[15:0]
  g1[2] = (int)((((unsigned)tensorW >> 16) & 0xFFFFu)
              | (((unsigned)tensorH & 0xFFFFu) << 16));         // dim0[31:16] | dim1[15:0]
  g1[3] = (int)((((unsigned)tensorH >> 16) & 0xFFFFu)
              | (32u << 16));                                   // dim1[31:16] | tile_dim0=32
  g1[4] = (int)128u;                                            // tile_dim1=128, tile_dim2=0
  g1[5] = (int)(unsigned)rowStrideElems;                        // tensor_dim0_stride[31:0]
  g1[6] = 0;                                                    // stride hi / dim1_stride lo
  g1[7] = 0;
  v4i z4 = {0, 0, 0, 0};
#if defined(__clang_major__) && (__clang_major__ >= 23)
  v8i z8 = {0, 0, 0, 0, 0, 0, 0, 0};
  __builtin_amdgcn_tensor_load_to_lds(g0, g1, z4, z4, z8, 0);
#else
  __builtin_amdgcn_tensor_load_to_lds(g0, g1, z4, z4, 0);
#endif
}
#endif

// ---------------------------------------------------------------------------
// Generic f16 GEMM with f32 accumulate: C = A[M,K] * B[K,N] (+bias)(ReLU)
// 8 waves, 128x128 block tile, BK=32; each wave: 32x64 output = 8 WMMAs/step.
// A tile staged by the Tensor Data Mover (wave 0), B transposed manually.
// ---------------------------------------------------------------------------
__global__ __launch_bounds__(256) void gemm_f16_wmma(
    const _Float16* __restrict__ A, const _Float16* __restrict__ Bm,
    const float* __restrict__ bias, float* __restrict__ Cf,
    _Float16* __restrict__ Ch, int M, int N, int K, int relu)
{
  __shared__ _Float16 As[128 * 40];   // [row 128][K 32] pitch 40 halves
  __shared__ _Float16 Bs[128 * 40];   // transposed: [N 128][K 32] pitch 40
  const int tid = threadIdx.x, lane = tid & 31, wid = tid >> 5;
  const int wr = wid & 3, wc = wid >> 2;             // 4 row-tiles x 2 col-halves
  const int m0 = blockIdx.y << 7, n0 = blockIdx.x << 7;
  v8f acc[2][4];
#pragma unroll
  for (int i = 0; i < 2; ++i)
#pragma unroll
    for (int j = 0; j < 4; ++j) acc[i][j] = (v8f){0,0,0,0,0,0,0,0};

  for (int ks = 0; ks < K; ks += 32) {
#if USE_TDM
    if (wid == 0) {
      tdm_load_tile_a(A + (size_t)m0 * K + ks,
                      (unsigned)(unsigned long long)(const void*)&As[0],
                      K, K, M);
    }
#else
#pragma unroll
    for (int h2 = 0; h2 < 2; ++h2) {
      const int idx = tid + h2 * 256;
      const int row = idx >> 2, seg = idx & 3;
      *(float4*)&As[row * 40 + seg * 8] =
          *(const float4*)(A + (size_t)(m0 + row) * K + ks + seg * 8);
    }
#endif
#pragma unroll
    for (int h2 = 0; h2 < 2; ++h2) {
      const int idx = tid + h2 * 256;
      const int kk = idx & 31, ng = (idx >> 5) << 3;
      float4 bv = *(const float4*)(Bm + (size_t)(ks + kk) * N + n0 + ng);
      const _Float16* hp = (const _Float16*)&bv;
#pragma unroll
      for (int i = 0; i < 8; ++i) Bs[(ng + i) * 40 + kk] = hp[i];
      if (ks + 32 < K)   // global_prefetch_b8 for next K-slab of B
        __builtin_prefetch(Bm + (size_t)(ks + 32 + kk) * N + n0 + ng, 0, 1);
    }
#if USE_TDM
    if (wid == 0) __builtin_amdgcn_s_wait_tensorcnt(0);
#endif
    __syncthreads();

    v16h a0 = load_frag(&As[(wr * 32 + (lane & 15)) * 40], lane);
    v16h a1 = load_frag(&As[(wr * 32 + 16 + (lane & 15)) * 40], lane);
#pragma unroll
    for (int bt = 0; bt < 4; ++bt) {
      v16h bf = load_frag(&Bs[(wc * 64 + bt * 16 + (lane & 15)) * 40], lane);
      acc[0][bt] = __builtin_amdgcn_wmma_f32_16x16x32_f16(false, a0, false, bf, (short)0, acc[0][bt], false, false);
      acc[1][bt] = __builtin_amdgcn_wmma_f32_16x16x32_f16(false, a1, false, bf, (short)0, acc[1][bt], false, false);
    }
    __syncthreads();
  }

  const int g = lane >> 4, nl = lane & 15;
#pragma unroll
  for (int at = 0; at < 2; ++at)
#pragma unroll
    for (int bt = 0; bt < 4; ++bt) {
      const int col = n0 + wc * 64 + bt * 16 + nl;
      const float bb = bias ? bias[col] : 0.0f;
#pragma unroll
      for (int r = 0; r < 8; ++r) {
        const int row = m0 + wr * 32 + at * 16 + g * 8 + r;
        float vv = acc[at][bt][r] + bb;
        if (relu) vv = fmaxf(vv, 0.0f);
        if (Ch) Ch[(size_t)row * N + col] = (_Float16)vv;
        else    Cf[(size_t)row * N + col] = vv;
      }
    }
}

// ---------------------------------------------------------------------------
// Flash attention: one wave per (b, h, 16-query block). 32-key superblocks so
// the P*V WMMA has exactly K=32. Online softmax with intra-half shuffles.
// ---------------------------------------------------------------------------
__global__ __launch_bounds__(32) void attn_flash(
    const _Float16* __restrict__ q, const _Float16* __restrict__ k,
    const _Float16* __restrict__ v, _Float16* __restrict__ o)
{
  __shared__ _Float16 Kt[32 * 72];   // [key 32][d 64] pitch 72 (16B-aligned rows)
  __shared__ _Float16 Vt[64 * 36];   // transposed: [d 64][key 32] pitch 36
  __shared__ _Float16 Pt[16 * 36];   // P tile f16: [qrow 16][key 32] pitch 36
  const int lane = threadIdx.x;
  const int qb = blockIdx.x, bh = blockIdx.y;
  const int b = bh / H_, h = bh % H_;
  const int q0 = qb << 4;
  const size_t tokbase = (size_t)b * T_;
  const int g = lane >> 4, nl = lane & 15;

  const _Float16* qrow = q + (tokbase + q0 + nl) * E_ + h * HD_;
  v16h qa0 = load_frag(qrow, lane);        // d 0..31
  v16h qa1 = load_frag(qrow + 32, lane);   // d 32..63

  float mrow[8], lrow[8];
  v8f acc[4];
#pragma unroll
  for (int r = 0; r < 8; ++r) { mrow[r] = -3.0e38f; lrow[r] = 0.0f; }
#pragma unroll
  for (int d = 0; d < 4; ++d) acc[d] = (v8f){0,0,0,0,0,0,0,0};
  const float scale = 0.125f;              // HD^-0.5

  const int nkb = (q0 + 15) / 32 + 1;
  for (int kb = 0; kb < nkb; ++kb) {
    const int k0 = kb << 5;
    const _Float16* ksrc = k + (tokbase + k0 + lane) * E_ + h * HD_;
    const _Float16* vsrc = v + (tokbase + k0 + lane) * E_ + h * HD_;
#pragma unroll
    for (int i = 0; i < 8; ++i)
      *(float4*)&Kt[lane * 72 + i * 8] = *(const float4*)(ksrc + i * 8);
    float4 vbuf[8];
#pragma unroll
    for (int i = 0; i < 8; ++i) vbuf[i] = *(const float4*)(vsrc + i * 8);
    const _Float16* vh = (const _Float16*)vbuf;
#pragma unroll
    for (int d = 0; d < 64; ++d) Vt[d * 36 + lane] = vh[d];   // transpose
    __syncthreads();

    // S = Q * K^T : two 16x16 tiles, K-dim = 64 (2 WMMAs each)
    v8f s0 = {0,0,0,0,0,0,0,0}, s1 = {0,0,0,0,0,0,0,0};
    v16h kb0  = load_frag(&Kt[nl * 72], lane);
    v16h kb0b = load_frag(&Kt[nl * 72 + 32], lane);
    v16h kb1  = load_frag(&Kt[(16 + nl) * 72], lane);
    v16h kb1b = load_frag(&Kt[(16 + nl) * 72 + 32], lane);
    s0 = __builtin_amdgcn_wmma_f32_16x16x32_f16(false, qa0, false, kb0,  (short)0, s0, false, false);
    s0 = __builtin_amdgcn_wmma_f32_16x16x32_f16(false, qa1, false, kb0b, (short)0, s0, false, false);
    s1 = __builtin_amdgcn_wmma_f32_16x16x32_f16(false, qa0, false, kb1,  (short)0, s1, false, false);
    s1 = __builtin_amdgcn_wmma_f32_16x16x32_f16(false, qa1, false, kb1b, (short)0, s1, false, false);

    // causal mask + online softmax (rows live in 16-lane halves)
#pragma unroll
    for (int r = 0; r < 8; ++r) {
      const int qi = q0 + g * 8 + r;
      float x0 = s0[r] * scale; if (k0 + nl > qi)      x0 = -3.0e38f;
      float x1 = s1[r] * scale; if (k0 + 16 + nl > qi) x1 = -3.0e38f;
      const float mnew = fmaxf(mrow[r], hmax16(fmaxf(x0, x1)));
      const float al = __expf(mrow[r] - mnew);
      const float p0 = __expf(x0 - mnew);
      const float p1 = __expf(x1 - mnew);
      mrow[r] = mnew;
      lrow[r] = lrow[r] * al + hsum16(p0 + p1);
      acc[0][r] *= al; acc[1][r] *= al; acc[2][r] *= al; acc[3][r] *= al;
      Pt[(g * 8 + r) * 36 + nl]      = (_Float16)p0;
      Pt[(g * 8 + r) * 36 + 16 + nl] = (_Float16)p1;
    }
    __syncthreads();

    // O += P(16x32) * V(32x64): 4 WMMAs (one per 16-wide d group)
    v16h pa = load_frag(&Pt[nl * 36], lane);
#pragma unroll
    for (int dg = 0; dg < 4; ++dg) {
      v16h vb = load_frag(&Vt[(dg * 16 + nl) * 36], lane);
      acc[dg] = __builtin_amdgcn_wmma_f32_16x16x32_f16(false, pa, false, vb, (short)0, acc[dg], false, false);
    }
    __syncthreads();
  }

#pragma unroll
  for (int dg = 0; dg < 4; ++dg)
#pragma unroll
    for (int r = 0; r < 8; ++r) {
      const int row = q0 + g * 8 + r;
      o[(tokbase + row) * E_ + h * HD_ + dg * 16 + nl] = (_Float16)(acc[dg][r] / lrow[r]);
    }
}

// ---------------------------------------------------------------------------
// x = LN(x + y) (y nullable), also emits f16 copy for next GEMM.
// ---------------------------------------------------------------------------
__global__ __launch_bounds__(256) void resid_ln(
    float* __restrict__ x, const float* __restrict__ y,
    const float* __restrict__ gw, const float* __restrict__ bw,
    _Float16* __restrict__ xh)
{
  __shared__ float ssum[256], ssq[256];
  const int tid = threadIdx.x;
  const size_t base = (size_t)blockIdx.x * E_;
  float loc[4]; float s = 0.0f, q = 0.0f;
#pragma unroll
  for (int i = 0; i < 4; ++i) {
    const int e = tid + i * 256;
    const float vv = x[base + e] + (y ? y[base + e] : 0.0f);
    loc[i] = vv; s += vv; q += vv * vv;
  }
  ssum[tid] = s; ssq[tid] = q; __syncthreads();
  for (int st = 128; st > 0; st >>= 1) {
    if (tid < st) { ssum[tid] += ssum[tid + st]; ssq[tid] += ssq[tid + st]; }
    __syncthreads();
  }
  const float mean = ssum[0] * (1.0f / E_);
  const float var  = ssq[0] * (1.0f / E_) - mean * mean;
  const float rstd = rsqrtf(var + 1e-5f);
#pragma unroll
  for (int i = 0; i < 4; ++i) {
    const int e = tid + i * 256;
    const float nv = (loc[i] - mean) * rstd * gw[e] + bw[e];
    x[base + e] = nv; xh[base + e] = (_Float16)nv;
  }
}

__global__ __launch_bounds__(256) void embed_gather(
    const int* __restrict__ idx, const float* __restrict__ emb,
    float* __restrict__ x, _Float16* __restrict__ xh)
{
  const int row = blockIdx.x;
  const int id = idx[row];
  const size_t src = (size_t)id * E_, dst = (size_t)row * E_;
#pragma unroll
  for (int i = 0; i < 4; ++i) {
    const int e = threadIdx.x + i * 256;
    const float vv = emb[src + e];
    x[dst + e] = vv; xh[dst + e] = (_Float16)vv;
  }
}

__global__ __launch_bounds__(256) void cast_f32_f16(
    const float* __restrict__ in, _Float16* __restrict__ out, size_t n)
{
  size_t i = (size_t)blockIdx.x * blockDim.x + threadIdx.x;
  const size_t stride = (size_t)gridDim.x * blockDim.x;
  for (; i < n; i += stride) out[i] = (_Float16)in[i];
}

// [L,H,E,HD] -> f16 [L, E, H*HD]
__global__ __launch_bounds__(256) void qkv_permute(
    const float* __restrict__ in, _Float16* __restrict__ out, size_t n)
{
  size_t i = (size_t)blockIdx.x * blockDim.x + threadIdx.x;
  const size_t stride = (size_t)gridDim.x * blockDim.x;
  for (; i < n; i += stride) {
    const int d = (int)(i % HD_);
    size_t t = i / HD_;
    const int e = (int)(t % E_); t /= E_;
    const int h = (int)(t % H_);
    const int l = (int)(t / H_);
    out[((size_t)l * E_ + e) * E_ + h * HD_ + d] = (_Float16)in[i];
  }
}

// ---------------------------------------------------------------------------
extern "C" void kernel_launch(void* const* d_in, const int* in_sizes, int n_in,
                              void* d_out, int out_size, void* d_ws, size_t ws_size,
                              hipStream_t stream) {
  (void)in_sizes; (void)n_in; (void)out_size; (void)ws_size;
  const int*   idx  = (const int*)d_in[0];
  const float* emb  = (const float*)d_in[1];
  const float* Wq   = (const float*)d_in[2];
  const float* Wk   = (const float*)d_in[3];
  const float* Wv   = (const float*)d_in[4];
  const float* Wo   = (const float*)d_in[5];
  const float* bo   = (const float*)d_in[6];
  const float* ln1g = (const float*)d_in[7];
  const float* ln1b = (const float*)d_in[8];
  const float* W1   = (const float*)d_in[9];
  const float* b1   = (const float*)d_in[10];
  const float* W2   = (const float*)d_in[11];
  const float* b2   = (const float*)d_in[12];
  const float* ln2g = (const float*)d_in[13];
  const float* ln2b = (const float*)d_in[14];
  const float* lnfg = (const float*)d_in[15];
  const float* lnfb = (const float*)d_in[16];
  const float* Wlm  = (const float*)d_in[17];
  const float* blm  = (const float*)d_in[18];
  float* out = (float*)d_out;

  // Workspace bump allocator (256B aligned). Total ~240 MB.
  size_t off = 0;
  auto alloc = [&](size_t bytes) -> void* {
    void* p = (char*)d_ws + off;
    off += (bytes + 255) & ~(size_t)255;
    return p;
  };
  float*    xf  = (float*)   alloc((size_t)NTOK * E_ * 4);
  float*    yf  = (float*)   alloc((size_t)NTOK * E_ * 4);
  _Float16* xh  = (_Float16*)alloc((size_t)NTOK * E_ * 2);
  _Float16* qh  = (_Float16*)alloc((size_t)NTOK * E_ * 2);
  _Float16* kh  = (_Float16*)alloc((size_t)NTOK * E_ * 2);
  _Float16* vh  = (_Float16*)alloc((size_t)NTOK * E_ * 2);
  _Float16* oh  = (_Float16*)alloc((size_t)NTOK * E_ * 2);
  _Float16* h1h = (_Float16*)alloc((size_t)NTOK * FF_ * 2);
  _Float16* WqH = (_Float16*)alloc((size_t)L_ * E_ * E_ * 2);
  _Float16* WkH = (_Float16*)alloc((size_t)L_ * E_ * E_ * 2);
  _Float16* WvH = (_Float16*)alloc((size_t)L_ * E_ * E_ * 2);
  _Float16* WoH = (_Float16*)alloc((size_t)L_ * E_ * E_ * 2);
  _Float16* W1H = (_Float16*)alloc((size_t)L_ * E_ * FF_ * 2);
  _Float16* W2H = (_Float16*)alloc((size_t)L_ * FF_ * E_ * 2);
  _Float16* WlmH= (_Float16*)alloc((size_t)E_ * V_ * 2);

  // --- weight prep ---
  const size_t nqkv = (size_t)L_ * H_ * E_ * HD_;
  qkv_permute<<<4096, 256, 0, stream>>>(Wq, WqH, nqkv);
  qkv_permute<<<4096, 256, 0, stream>>>(Wk, WkH, nqkv);
  qkv_permute<<<4096, 256, 0, stream>>>(Wv, WvH, nqkv);
  cast_f32_f16<<<4096, 256, 0, stream>>>(Wo, WoH, (size_t)L_ * E_ * E_);
  cast_f32_f16<<<4096, 256, 0, stream>>>(W1, W1H, (size_t)L_ * E_ * FF_);
  cast_f32_f16<<<4096, 256, 0, stream>>>(W2, W2H, (size_t)L_ * FF_ * E_);
  cast_f32_f16<<<4096, 256, 0, stream>>>(Wlm, WlmH, (size_t)E_ * V_);

  auto gemm = [&](const _Float16* A, const _Float16* Bm, const float* bias,
                  float* Cf, _Float16* Ch, int M, int N, int K, int relu) {
    dim3 g(N / 128, M / 128);
    gemm_f16_wmma<<<g, 256, 0, stream>>>(A, Bm, bias, Cf, Ch, M, N, K, relu);
  };

  // --- embedding ---
  embed_gather<<<NTOK, 256, 0, stream>>>(idx, emb, xf, xh);

  // --- transformer blocks ---
  for (int l = 0; l < L_; ++l) {
    const _Float16* wq = WqH + (size_t)l * E_ * E_;
    const _Float16* wk = WkH + (size_t)l * E_ * E_;
    const _Float16* wv = WvH + (size_t)l * E_ * E_;
    const _Float16* wo = WoH + (size_t)l * E_ * E_;
    const _Float16* w1 = W1H + (size_t)l * E_ * FF_;
    const _Float16* w2 = W2H + (size_t)l * FF_ * E_;

    gemm(xh, wq, nullptr, nullptr, qh, NTOK, E_, E_, 0);
    gemm(xh, wk, nullptr, nullptr, kh, NTOK, E_, E_, 0);
    gemm(xh, wv, nullptr, nullptr, vh, NTOK, E_, E_, 0);
    attn_flash<<<dim3(T_ / 16, B_ * H_), 32, 0, stream>>>(qh, kh, vh, oh);
    gemm(oh, wo, bo + (size_t)l * E_, yf, nullptr, NTOK, E_, E_, 0);
    resid_ln<<<NTOK, 256, 0, stream>>>(xf, yf, ln1g + (size_t)l * E_, ln1b + (size_t)l * E_, xh);
    gemm(xh, w1, b1 + (size_t)l * FF_, nullptr, h1h, NTOK, FF_, E_, 1);
    gemm(h1h, w2, b2 + (size_t)l * E_, yf, nullptr, NTOK, E_, FF_, 0);
    resid_ln<<<NTOK, 256, 0, stream>>>(xf, yf, ln2g + (size_t)l * E_, ln2b + (size_t)l * E_, xh);
  }

  // --- final LN + LM head ---
  resid_ln<<<NTOK, 256, 0, stream>>>(xf, nullptr, lnfg, lnfb, xh);
  gemm(xh, WlmH, blm, out, nullptr, NTOK, V_, E_, 0);
}
